// Linear_85005992722465
// MI455X (gfx1250) — matrix operand order repaired and checked by reading, not compiled, not run on hardware
//
#include <hip/hip_runtime.h>
#include <hip/hip_bf16.h>

// ---------------------------------------------------------------------------
// Block-diagonal Cayley/Neumann rotation apply for MI455X (gfx1250, wave32).
//
// Stage 1: build_rot  -- per 64x64 block r: R = I + 2Q + 2Q^2 + 2Q^3 + 2Q^4
//                        (Q skew-symmetric from packed triu vector), store
//                        R^T in f16 at ws[r][c][k] (WMMA-B-friendly layout).
// Stage 2: apply_rot  -- out[m, r*64+c] = sum_k x[m, r*64+k] * R[k][c]
//                        via v_wmma_f32_16x16x32_f16 (f16 in, f32 accumulate).
// ---------------------------------------------------------------------------

typedef __attribute__((ext_vector_type(16))) _Float16 v16h;
typedef __attribute__((ext_vector_type(8)))  float    v8f;
typedef __attribute__((ext_vector_type(4)))  float    v4f;

#define R_BLOCKS   64
#define BS         64          // block size
#define N_ELEM     2016        // 64*63/2
#define LDQ        68          // LDS row stride (floats): 16B aligned, conflict-free
#define LDX        4096        // x / out row stride (floats) = R_BLOCKS*BS

// ------------------------- Stage 1: build rotations -------------------------
// One 256-thread workgroup per block r. Thread t owns row i = t/4,
// columns j0..j0+15 with j0 = (t%4)*16. Series accumulator kept in registers.
__global__ __launch_bounds__(256) void build_rot(const float* __restrict__ w,
                                                 _Float16* __restrict__ rotT) {
    __shared__ float Qs [BS * LDQ];
    __shared__ float Pas[BS * LDQ];
    __shared__ float Pbs[BS * LDQ];

    const int r  = blockIdx.x;
    const int t  = threadIdx.x;
    const int i  = t >> 2;          // row 0..63
    const int j0 = (t & 3) << 4;    // col base 0,16,32,48

    const float* wr = w + r * N_ELEM;

    float racc[16];
    #pragma unroll
    for (int j = 0; j < 16; ++j) {
        const int col = j0 + j;
        float qv = 0.0f;
        if (col > i) {
            // triu index for (i, col), row-major, k=1 diagonal offset
            qv =  wr[i * 63 - (i * (i - 1)) / 2 + (col - i - 1)];
        } else if (col < i) {
            qv = -wr[col * 63 - (col * (col - 1)) / 2 + (i - col - 1)];
        }
        Qs[i * LDQ + col] = qv;
        racc[j] = ((col == i) ? 1.0f : 0.0f) + 2.0f * qv;   // I + 2Q
    }
    __syncthreads();

    // P <- Q; three times: P <- P @ Q, racc += 2*P   (gives Q^2, Q^3, Q^4)
    float* src = Qs;
    float* dst = Pas;
    for (int pw = 0; pw < 3; ++pw) {
        float acc[16];
        #pragma unroll
        for (int j = 0; j < 16; ++j) acc[j] = 0.0f;

        const v4f* qrow_base = (const v4f*)(Qs + j0);   // 16B aligned (j0 mult of 16)
        for (int k = 0; k < BS; ++k) {
            const float p = src[i * LDQ + k];
            const v4f* qk = (const v4f*)((const float*)qrow_base + k * LDQ);
            v4f q0 = qk[0], q1 = qk[1], q2 = qk[2], q3 = qk[3];
            #pragma unroll
            for (int j = 0; j < 4; ++j) {
                acc[j +  0] = fmaf(p, q0[j], acc[j +  0]);
                acc[j +  4] = fmaf(p, q1[j], acc[j +  4]);
                acc[j +  8] = fmaf(p, q2[j], acc[j +  8]);
                acc[j + 12] = fmaf(p, q3[j], acc[j + 12]);
            }
        }
        #pragma unroll
        for (int j = 0; j < 16; ++j) {
            dst[i * LDQ + j0 + j] = acc[j];
            racc[j] += 2.0f * acc[j];
        }
        __syncthreads();
        src = dst;
        dst = (src == Pas) ? Pbs : Pas;
    }

    // Store R^T in f16: rotT[r][c][k] = R[k][c].  Thread writes k-row i,
    // columns j0..j0+15 -> rotT[r*4096 + (j0+j)*64 + i].
    _Float16* ob = rotT + (size_t)r * (BS * BS);
    #pragma unroll
    for (int j = 0; j < 16; ++j)
        ob[(j0 + j) * BS + i] = (_Float16)racc[j];
}

// --------------------------- Stage 2: apply (WMMA) --------------------------
// Grid: 512 workgroups = 8 M-slabs x 64 blocks; 256 threads = 8 wave32 each.
// Each wave processes 8 strips of 16 rows; per strip: 4 N-tiles x 2 K-steps
// of v_wmma_f32_16x16x32_f16 against B fragments held in registers.
#define STRIPS     8
#define M_PER_WG   1024         // 8 waves * STRIPS * 16 rows

__global__ __launch_bounds__(256) void apply_rot(const float* __restrict__ x,
                                                 const _Float16* __restrict__ rotT,
                                                 float* __restrict__ out) {
    const int r    = blockIdx.x & (R_BLOCKS - 1);
    const int slab = blockIdx.x >> 6;
    const int wave = threadIdx.x >> 5;
    const int lane = threadIdx.x & 31;
    const int half = lane >> 4;     // 0: lanes 0-15, 1: lanes 16-31
    const int lrow = lane & 15;

    // --- B fragments (rot^T), 8x v16h, loaded once, reused over all strips.
    // B order used here (16-bit, 32x16), not the ISA table order: lanes 0-15 carry
    // K=0..15, lanes 16-31 carry K=16..31; N = lane%16.  With rotT stored as
    // [c][k], element h of the fragment = rotT[(n0+lrow)*64 + kt + half*16 + h]
    // -> 16 contiguous halfs = one aligned 32B load.
    v16h bf[8];
    const _Float16* rb = rotT + (size_t)r * (BS * BS);
    #pragma unroll
    for (int nt = 0; nt < 4; ++nt)
        #pragma unroll
        for (int kt = 0; kt < 2; ++kt)
            bf[nt * 2 + kt] =
                *(const v16h*)(rb + (nt * 16 + lrow) * BS + kt * 32 + half * 16);

    const int col0 = r * BS;

    for (int it = 0; it < STRIPS; ++it) {
        const int m0 = slab * M_PER_WG + (it * 8 + wave) * 16;
        const int m  = m0 + lrow;
        const float* ab = x + (size_t)m * LDX + col0;

        if (it + 1 < STRIPS)   // pull next strip toward the caches (global_prefetch)
            __builtin_prefetch(ab + (size_t)(8 * 16) * LDX, 0, 1);

        // --- A fragments (16x32): lane<16 holds K 0..15, lane>=16 holds
        // K 16..31 for its row (matches B's contiguous K order).
        // Each half is one contiguous float16 run -> 4x nontemporal v4f loads.
        v16h af[2];
        #pragma unroll
        for (int kt = 0; kt < 2; ++kt) {
            const float* p0 = ab + kt * 32 + half * 16;       // K run 1 (8 floats)
            const float* p1 = p0 + 8;                         // K run 2 (8 floats)
            v4f f0 = __builtin_nontemporal_load((const v4f*)p0);
            v4f f1 = __builtin_nontemporal_load((const v4f*)(p0 + 4));
            v4f f2 = __builtin_nontemporal_load((const v4f*)p1);
            v4f f3 = __builtin_nontemporal_load((const v4f*)(p1 + 4));
            v16h a;
            #pragma unroll
            for (int j = 0; j < 4; ++j) {
                a[j +  0] = (_Float16)f0[j];
                a[j +  4] = (_Float16)f1[j];
                a[j +  8] = (_Float16)f2[j];
                a[j + 12] = (_Float16)f3[j];
            }
            af[kt] = a;
        }

        // --- 8 WMMAs: C[nt] += A(k0..31) * B + A(k32..63) * B
        v8f c0 = {}, c1 = {}, c2 = {}, c3 = {};
        c0 = __builtin_amdgcn_wmma_f32_16x16x32_f16(false, af[0], false, bf[0], (short)0, c0, false, false);
        c0 = __builtin_amdgcn_wmma_f32_16x16x32_f16(false, af[1], false, bf[1], (short)0, c0, false, false);
        c1 = __builtin_amdgcn_wmma_f32_16x16x32_f16(false, af[0], false, bf[2], (short)0, c1, false, false);
        c1 = __builtin_amdgcn_wmma_f32_16x16x32_f16(false, af[1], false, bf[3], (short)0, c1, false, false);
        c2 = __builtin_amdgcn_wmma_f32_16x16x32_f16(false, af[0], false, bf[4], (short)0, c2, false, false);
        c2 = __builtin_amdgcn_wmma_f32_16x16x32_f16(false, af[1], false, bf[5], (short)0, c2, false, false);
        c3 = __builtin_amdgcn_wmma_f32_16x16x32_f16(false, af[0], false, bf[6], (short)0, c3, false, false);
        c3 = __builtin_amdgcn_wmma_f32_16x16x32_f16(false, af[1], false, bf[7], (short)0, c3, false, false);

        // --- Store: C/D layout: VGPR i -> row M = half*8 + i, col N = lrow.
        float* ob = out + (size_t)(m0 + half * 8) * LDX + col0 + lrow;
        #pragma unroll
        for (int i = 0; i < 8; ++i) {
            float* row = ob + (size_t)i * LDX;
            __builtin_nontemporal_store(c0[i], row +  0);
            __builtin_nontemporal_store(c1[i], row + 16);
            __builtin_nontemporal_store(c2[i], row + 32);
            __builtin_nontemporal_store(c3[i], row + 48);
        }
    }
}

// ------------------------------- launcher -----------------------------------
extern "C" void kernel_launch(void* const* d_in, const int* in_sizes, int n_in,
                              void* d_out, int out_size, void* d_ws, size_t ws_size,
                              hipStream_t stream) {
    const float* x = (const float*)d_in[0];   // (4, 2048, 4096) f32
    const float* w = (const float*)d_in[1];   // (64, 2016) f32
    float*     out = (float*)d_out;           // (4, 2048, 4096) f32
    _Float16* rotT = (_Float16*)d_ws;         // 64*64*64 halfs = 512 KB

    build_rot<<<R_BLOCKS, 256, 0, stream>>>(w, rotT);

    const int m_total = 4 * 2048;             // 8192 rows
    const int grid    = (m_total / M_PER_WG) * R_BLOCKS;  // 8 * 64 = 512
    apply_rot<<<grid, 256, 0, stream>>>(x, rotT, out);
}
